// DrugCombinationPredictor_89197880803532
// MI455X (gfx1250) — matrix operand order repaired and checked by reading, not compiled, not run on hardware
//
#include <hip/hip_runtime.h>
#include <hip/hip_bf16.h>
#include <math.h>

typedef __bf16 bf16;
typedef __attribute__((ext_vector_type(16))) __bf16 v16bf;
typedef __attribute__((ext_vector_type(8)))  __bf16 v8bf;
typedef __attribute__((ext_vector_type(8)))  float  v8f;

#define NPG_   40
#define EPG_   160
#define ETOT_  (EPG_ + NPG_)          // edges + self loops per graph
#define NGRAPH 4096
#define NNODES (NGRAPH * NPG_)        // 163840
#define ETOTAL (NGRAPH * EPG_)        // 655360
#define BN_EPS_ 1e-5f

static __host__ int cdiv(int a, int b) { return (a + b - 1) / b; }

// ---------------------------------------------------------------------------
// f32 [rows,K] -> bf16 [rows,Kpad] (zero padded)
__global__ void cvt_a_bf16(const float* __restrict__ X, bf16* __restrict__ Xb,
                           int rows, int K, int Kpad) {
  int idx = blockIdx.x * blockDim.x + threadIdx.x;
  if (idx >= rows * Kpad) return;
  int r = idx / Kpad, k = idx - r * Kpad;
  Xb[idx] = (k < K) ? (bf16)X[(size_t)r * K + k] : (bf16)0.0f;
}

// f32 W[K,D] -> bf16 Wt[D,Kpad] (transposed, zero padded)
__global__ void cvt_w_bf16_t(const float* __restrict__ W, bf16* __restrict__ Wt,
                             int K, int D, int Kpad) {
  int idx = blockIdx.x * blockDim.x + threadIdx.x;
  if (idx >= D * Kpad) return;
  int d = idx / Kpad, k = idx - d * Kpad;
  Wt[idx] = (k < K) ? (bf16)W[(size_t)k * D + d] : (bf16)0.0f;
}

// ---------------------------------------------------------------------------
// Build a 16-bit A fragment in this code's contiguous k order (B uses the same):
// lane&15 = row; lane>>4 selects K halves {0..15} vs {16..31}.
static __device__ __forceinline__ v16bf load_a_frag(const bf16* p) {
  v8bf lo = *(const v8bf*)(p);        // K = base + 0..7 (of this lane group)
  v8bf hi = *(const v8bf*)(p + 8);    // K = base + 8..15
  return __builtin_shufflevector(lo, hi, 0,1,2,3,4,5,6,7,8,9,10,11,12,13,14,15);
}

// C[M,Nd] = A[M,Ka] * B with B supplied transposed as Bt[Nd,Ka] (bf16).
// One wave -> 32x64 output tile: 2 A fragments x 4 B fragments = 8 WMMAs/K-step.
// Each B fragment is reused by 2 WMMAs (halves L2 weight traffic vs 16x64).
__global__ void __launch_bounds__(256)
gemm_wmma_bf16(const bf16* __restrict__ A, const bf16* __restrict__ Bt,
               float* __restrict__ C, int M, int Nd, int Ka, int total_waves) {
  int gwave = (blockIdx.x * blockDim.x + threadIdx.x) >> 5;
  if (gwave >= total_waves) return;
  int lane = threadIdx.x & 31;
  int ntn = Nd >> 6;
  int tm = gwave / ntn;
  int tn = gwave - tm * ntn;
  int mloc = lane & 15;
  int hi = lane >> 4;                       // lane group selects K-halves
  const bf16* arow0 = A + (size_t)(tm * 32 + mloc) * Ka + hi * 16;
  const bf16* arow1 = arow0 + (size_t)16 * Ka;
  int nbase = tn * 64 + (lane & 15);
  const bf16* b0 = Bt + (size_t)(nbase +  0) * Ka + hi * 16;
  const bf16* b1 = Bt + (size_t)(nbase + 16) * Ka + hi * 16;
  const bf16* b2 = Bt + (size_t)(nbase + 32) * Ka + hi * 16;
  const bf16* b3 = Bt + (size_t)(nbase + 48) * Ka + hi * 16;
  v8f acc0 = {}, acc1 = {}, acc2 = {}, acc3 = {};
  v8f acc4 = {}, acc5 = {}, acc6 = {}, acc7 = {};
  for (int k0 = 0; k0 < Ka; k0 += 32) {
    __builtin_prefetch(arow0 + k0 + 128, 0, 1);   // global_prefetch_b8, A stream
    __builtin_prefetch(arow1 + k0 + 128, 0, 1);
    v16bf a0 = load_a_frag(arow0 + k0);
    v16bf a1 = load_a_frag(arow1 + k0);
    v16bf vb0 = *(const v16bf*)(b0 + k0);
    v16bf vb1 = *(const v16bf*)(b1 + k0);
    v16bf vb2 = *(const v16bf*)(b2 + k0);
    v16bf vb3 = *(const v16bf*)(b3 + k0);
    acc0 = __builtin_amdgcn_wmma_f32_16x16x32_bf16(false, a0, false, vb0, (short)0, acc0, false, false);
    acc4 = __builtin_amdgcn_wmma_f32_16x16x32_bf16(false, a1, false, vb0, (short)0, acc4, false, false);
    acc1 = __builtin_amdgcn_wmma_f32_16x16x32_bf16(false, a0, false, vb1, (short)0, acc1, false, false);
    acc5 = __builtin_amdgcn_wmma_f32_16x16x32_bf16(false, a1, false, vb1, (short)0, acc5, false, false);
    acc2 = __builtin_amdgcn_wmma_f32_16x16x32_bf16(false, a0, false, vb2, (short)0, acc2, false, false);
    acc6 = __builtin_amdgcn_wmma_f32_16x16x32_bf16(false, a1, false, vb2, (short)0, acc6, false, false);
    acc3 = __builtin_amdgcn_wmma_f32_16x16x32_bf16(false, a0, false, vb3, (short)0, acc3, false, false);
    acc7 = __builtin_amdgcn_wmma_f32_16x16x32_bf16(false, a1, false, vb3, (short)0, acc7, false, false);
  }
  // C/D layout: VGPR r -> M = (lane>>4)*8 + r, N = lane&15
  int ncol = tn * 64 + (lane & 15);
  float* crow0 = C + (size_t)(tm * 32 + hi * 8) * Nd + ncol;
  float* crow1 = crow0 + (size_t)16 * Nd;
#pragma unroll
  for (int r = 0; r < 8; ++r) {
    crow0[(size_t)r * Nd +  0] = acc0[r];
    crow0[(size_t)r * Nd + 16] = acc1[r];
    crow0[(size_t)r * Nd + 32] = acc2[r];
    crow0[(size_t)r * Nd + 48] = acc3[r];
    crow1[(size_t)r * Nd +  0] = acc4[r];
    crow1[(size_t)r * Nd + 16] = acc5[r];
    crow1[(size_t)r * Nd + 32] = acc6[r];
    crow1[(size_t)r * Nd + 48] = acc7[r];
  }
}

// ---------------------------------------------------------------------------
// es[n,h] = sum_c H[n,h,c]*a_src[h,c]; ed likewise
__global__ void attn_scores(const float* __restrict__ H,
                            const float* __restrict__ as, const float* __restrict__ ad,
                            float* __restrict__ es, float* __restrict__ ed,
                            int Nn, int Hh, int Cc) {
  int idx = blockIdx.x * blockDim.x + threadIdx.x;
  if (idx >= Nn * Hh) return;
  int n = idx / Hh, h = idx - n * Hh;
  const float* hp = H + (size_t)n * (Hh * Cc) + h * Cc;
  const float* ap = as + h * Cc;
  const float* dp = ad + h * Cc;
  float s = 0.f, d = 0.f;
  for (int c = 0; c < Cc; ++c) { float v = hp[c]; s += v * ap[c]; d += v * dp[c]; }
  es[idx] = s; ed[idx] = d;
}

// ---------------------------------------------------------------------------
// One workgroup (256 threads) per graph: segment softmax (with self-loops) +
// alpha-weighted aggregation in LDS + bias + BN(+ReLU); writes bf16 (next GEMM
// operand) and/or f32.
__global__ void __launch_bounds__(256)
gat_aggregate(const float* __restrict__ H, const float* __restrict__ es,
              const float* __restrict__ ed, const int* __restrict__ edge,
              const float* __restrict__ bias,
              const float* __restrict__ bng, const float* __restrict__ bnb,
              const float* __restrict__ bnm, const float* __restrict__ bnv,
              bf16* __restrict__ out_bf, int out_kpad,
              float* __restrict__ out_f32,
              int Hh, int Cc, int relu_flag) {
  int g = blockIdx.x;
  int tid = threadIdx.x;
  __shared__ int   s_src[ETOT_], s_dst[ETOT_];
  __shared__ float s_es[NPG_ * 4], s_ed[NPG_ * 4];
  __shared__ float s_e[4][ETOT_];
  __shared__ float s_mx[NPG_ * 4], s_sm[NPG_ * 4];
  __shared__ float s_acc[NPG_ * 128];
  // 1) edges (+ self loops), local scores
  for (int j = tid; j < ETOT_; j += 256) {
    if (j < EPG_) {
      s_src[j] = edge[(size_t)g * EPG_ + j] - g * NPG_;
      s_dst[j] = edge[(size_t)ETOTAL + (size_t)g * EPG_ + j] - g * NPG_;
    } else {
      s_src[j] = j - EPG_; s_dst[j] = j - EPG_;
    }
  }
  for (int j = tid; j < NPG_ * Hh; j += 256) {
    s_es[j] = es[(size_t)g * NPG_ * Hh + j];
    s_ed[j] = ed[(size_t)g * NPG_ * Hh + j];
  }
  __syncthreads();
  // 2) raw attention logits with leaky_relu(0.2)
  for (int t = tid; t < ETOT_ * Hh; t += 256) {
    int e = t % ETOT_, h = t / ETOT_;
    float v = s_es[s_src[e] * Hh + h] + s_ed[s_dst[e] * Hh + h];
    s_e[h][e] = (v > 0.f) ? v : 0.2f * v;
  }
  __syncthreads();
  // 3) per (dst,head): max then sum of exp
  for (int t = tid; t < NPG_ * Hh; t += 256) {
    int i = t / Hh, h = t - i * Hh;
    float mx = -3.4e38f;
    for (int e = 0; e < ETOT_; ++e) if (s_dst[e] == i) mx = fmaxf(mx, s_e[h][e]);
    float sm = 0.f;
    for (int e = 0; e < ETOT_; ++e) if (s_dst[e] == i) sm += __expf(s_e[h][e] - mx);
    s_mx[i * Hh + h] = mx; s_sm[i * Hh + h] = sm;
  }
  __syncthreads();
  // 4) alpha (overwrite s_e)
  for (int t = tid; t < ETOT_ * Hh; t += 256) {
    int e = t % ETOT_, h = t / ETOT_;
    int d = s_dst[e];
    s_e[h][e] = __expf(s_e[h][e] - s_mx[d * Hh + h]) / s_sm[d * Hh + h];
  }
  __syncthreads();
  // 5) aggregate per head in LDS, fused epilogue
  int c = tid & (Cc - 1);
  int eo = tid / Cc;              // 2 edges in flight (Cc==128, 256 threads)
  for (int h = 0; h < Hh; ++h) {
    for (int t = tid; t < NPG_ * Cc; t += 256) s_acc[t] = 0.f;
    __syncthreads();
    for (int e = eo; e < ETOT_; e += 2) {
      float a = s_e[h][e];
      float v = H[((size_t)(g * NPG_ + s_src[e]) * Hh + h) * Cc + c] * a;
      atomicAdd(&s_acc[s_dst[e] * Cc + c], v);
    }
    __syncthreads();
    for (int t = tid; t < NPG_ * Cc; t += 256) {
      int i = t / Cc, c2 = t - i * Cc;
      int dch = h * Cc + c2;
      float y = s_acc[t] + bias[dch];
      y = (y - bnm[dch]) * rsqrtf(bnv[dch] + BN_EPS_) * bng[dch] + bnb[dch];
      if (relu_flag) y = fmaxf(y, 0.f);
      size_t node = (size_t)g * NPG_ + i;
      if (out_bf)  out_bf[node * out_kpad + dch] = (bf16)y;
      if (out_f32) out_f32[node * (size_t)(Hh * Cc) + dch] = y;
    }
    __syncthreads();
  }
}

// ---------------------------------------------------------------------------
// mean || max over the 40 nodes of each graph -> bf16 columns of combined vec
__global__ void pool_kernel(const float* __restrict__ H2, bf16* __restrict__ cmb,
                            int col_off) {
  int g = blockIdx.x, c = threadIdx.x;  // blockDim = 128
  float sm = 0.f, mx = -3.4e38f;
  for (int i = 0; i < NPG_; ++i) {
    float v = H2[((size_t)g * NPG_ + i) * 128 + c];
    sm += v; mx = fmaxf(mx, v);
  }
  cmb[(size_t)g * 544 + col_off + c]       = (bf16)(sm * (1.0f / NPG_));
  cmb[(size_t)g * 544 + col_off + 128 + c] = (bf16)mx;
}

__global__ void fill_addf(const float* __restrict__ addf, bf16* __restrict__ cmb) {
  int g = blockIdx.x, t = threadIdx.x;  // blockDim = 32 -> cols 512..543
  cmb[(size_t)g * 544 + 512 + t] = (t < 8) ? (bf16)addf[(size_t)g * 8 + t] : (bf16)0.0f;
}

// bias + BN + ReLU, dual-format output
__global__ void bn_relu_epi(const float* __restrict__ T, const float* __restrict__ bias,
                            const float* __restrict__ gg, const float* __restrict__ bb,
                            const float* __restrict__ mm, const float* __restrict__ vv,
                            bf16* __restrict__ out_bf, float* __restrict__ out_f32,
                            int D, int total) {
  int idx = blockIdx.x * blockDim.x + threadIdx.x;
  if (idx >= total) return;
  int d = idx % D;
  float y = T[idx] + bias[d];
  y = (y - mm[d]) * rsqrtf(vv[d] + BN_EPS_) * gg[d] + bb[d];
  y = fmaxf(y, 0.f);
  if (out_bf)  out_bf[idx] = (bf16)y;
  if (out_f32) out_f32[idx] = y;
}

__global__ void final_sigmoid(const float* __restrict__ h1, const float* __restrict__ W2,
                              const float* __restrict__ b2, float* __restrict__ out) {
  int g = blockIdx.x * blockDim.x + threadIdx.x;
  if (g >= NGRAPH) return;
  float s = b2[0];
  for (int c = 0; c < 128; ++c) s += h1[(size_t)g * 128 + c] * W2[c];
  out[g] = 1.f / (1.f + __expf(-s));
}

// ---------------------------------------------------------------------------
extern "C" void kernel_launch(void* const* d_in, const int* in_sizes, int n_in,
                              void* d_out, int out_size, void* d_ws, size_t ws_size,
                              hipStream_t stream) {
  (void)in_sizes; (void)n_in; (void)out_size; (void)ws_size;
  const float* x_a    = (const float*)d_in[0];
  const int*   edge_a = (const int*)  d_in[1];
  const float* x_b    = (const float*)d_in[3];
  const int*   edge_b = (const int*)  d_in[4];
  const float* addf   = (const float*)d_in[6];
  const float* gW0 = (const float*)d_in[7],  *gb0 = (const float*)d_in[8];
  const float* gas0= (const float*)d_in[9],  *gad0= (const float*)d_in[10];
  const float* gW1 = (const float*)d_in[11], *gb1 = (const float*)d_in[12];
  const float* gas1= (const float*)d_in[13], *gad1= (const float*)d_in[14];
  const float* gW2 = (const float*)d_in[15], *gb2 = (const float*)d_in[16];
  const float* gas2= (const float*)d_in[17], *gad2= (const float*)d_in[18];
  const float* bn0g=(const float*)d_in[19], *bn0b=(const float*)d_in[20];
  const float* bn0m=(const float*)d_in[21], *bn0v=(const float*)d_in[22];
  const float* bn1g=(const float*)d_in[23], *bn1b=(const float*)d_in[24];
  const float* bn1m=(const float*)d_in[25], *bn1v=(const float*)d_in[26];
  const float* bn2g=(const float*)d_in[27], *bn2b=(const float*)d_in[28];
  const float* bn2m=(const float*)d_in[29], *bn2v=(const float*)d_in[30];
  const float* mW0=(const float*)d_in[31], *mb0=(const float*)d_in[32];
  const float* mg0=(const float*)d_in[33], *mbe0=(const float*)d_in[34];
  const float* mm0=(const float*)d_in[35], *mv0=(const float*)d_in[36];
  const float* mW1=(const float*)d_in[37], *mb1=(const float*)d_in[38];
  const float* mg1=(const float*)d_in[39], *mbe1=(const float*)d_in[40];
  const float* mm1=(const float*)d_in[41], *mv1=(const float*)d_in[42];
  const float* mW2=(const float*)d_in[43], *mb2=(const float*)d_in[44];

  char* ws = (char*)d_ws;
  size_t off = 0;
  auto salloc = [&](size_t bytes) -> void* {
    void* p = ws + off; off += (bytes + 255) & ~(size_t)255; return p;
  };
  bf16*  Abf  = (bf16*) salloc((size_t)NNODES * 512 * 2);
  float* Hbuf = (float*)salloc((size_t)NNODES * 512 * 4);
  float* es   = (float*)salloc((size_t)NNODES * 4 * 4);
  float* ed   = (float*)salloc((size_t)NNODES * 4 * 4);
  float* H2   = (float*)salloc((size_t)NNODES * 128 * 4);
  bf16*  Cmb  = (bf16*) salloc((size_t)NGRAPH * 544 * 2);
  float* t0   = (float*)salloc((size_t)NGRAPH * 256 * 4);
  bf16*  h0b  = (bf16*) salloc((size_t)NGRAPH * 256 * 2);
  float* t1   = (float*)salloc((size_t)NGRAPH * 128 * 4);
  float* h1   = (float*)salloc((size_t)NGRAPH * 128 * 4);
  bf16*  W0t  = (bf16*) salloc((size_t)512 * 96 * 2);
  bf16*  W1t  = (bf16*) salloc((size_t)512 * 512 * 2);
  bf16*  W2t  = (bf16*) salloc((size_t)128 * 512 * 2);
  bf16*  MW0t = (bf16*) salloc((size_t)256 * 544 * 2);
  bf16*  MW1t = (bf16*) salloc((size_t)128 * 256 * 2);

  auto gemm = [&](const bf16* A, const bf16* Bt, float* C, int M, int Nd, int Ka) {
    int waves = (M / 32) * (Nd / 64);
    int blocks = cdiv(waves * 32, 256);
    gemm_wmma_bf16<<<blocks, 256, 0, stream>>>(A, Bt, C, M, Nd, Ka, waves);
  };

  // weight conversions (f32 -> bf16, transposed, K padded to /32)
  cvt_w_bf16_t<<<cdiv(512 * 96, 256), 256, 0, stream>>>(gW0, W0t, 78, 512, 96);
  cvt_w_bf16_t<<<cdiv(512 * 512, 256), 256, 0, stream>>>(gW1, W1t, 512, 512, 512);
  cvt_w_bf16_t<<<cdiv(128 * 512, 256), 256, 0, stream>>>(gW2, W2t, 512, 128, 512);
  cvt_w_bf16_t<<<cdiv(256 * 544, 256), 256, 0, stream>>>(mW0, MW0t, 520, 256, 544);
  cvt_w_bf16_t<<<cdiv(128 * 256, 256), 256, 0, stream>>>(mW1, MW1t, 256, 128, 256);

  auto encode = [&](const float* x, const int* edge, int col_off) {
    cvt_a_bf16<<<cdiv(NNODES * 96, 256), 256, 0, stream>>>(x, Abf, NNODES, 78, 96);
    // layer 0: heads=4, ch=128, concat, relu
    gemm(Abf, W0t, Hbuf, NNODES, 512, 96);
    attn_scores<<<cdiv(NNODES * 4, 256), 256, 0, stream>>>(Hbuf, gas0, gad0, es, ed, NNODES, 4, 128);
    gat_aggregate<<<NGRAPH, 256, 0, stream>>>(Hbuf, es, ed, edge, gb0,
        bn0g, bn0b, bn0m, bn0v, Abf, 512, (float*)nullptr, 4, 128, 1);
    // layer 1
    gemm(Abf, W1t, Hbuf, NNODES, 512, 512);
    attn_scores<<<cdiv(NNODES * 4, 256), 256, 0, stream>>>(Hbuf, gas1, gad1, es, ed, NNODES, 4, 128);
    gat_aggregate<<<NGRAPH, 256, 0, stream>>>(Hbuf, es, ed, edge, gb1,
        bn1g, bn1b, bn1m, bn1v, Abf, 512, (float*)nullptr, 4, 128, 1);
    // layer 2: heads=1, ch=128, no relu
    gemm(Abf, W2t, Hbuf, NNODES, 128, 512);
    attn_scores<<<cdiv(NNODES * 1, 256), 256, 0, stream>>>(Hbuf, gas2, gad2, es, ed, NNODES, 1, 128);
    gat_aggregate<<<NGRAPH, 256, 0, stream>>>(Hbuf, es, ed, edge, gb2,
        bn2g, bn2b, bn2m, bn2v, (bf16*)nullptr, 0, H2, 1, 128, 0);
    pool_kernel<<<NGRAPH, 128, 0, stream>>>(H2, Cmb, col_off);
  };

  encode(x_a, edge_a, 0);
  encode(x_b, edge_b, 256);
  fill_addf<<<NGRAPH, 32, 0, stream>>>(addf, Cmb);

  // combination MLP: 520(->544) -> 256 -> 128 -> 1
  gemm(Cmb, MW0t, t0, NGRAPH, 256, 544);
  bn_relu_epi<<<cdiv(NGRAPH * 256, 256), 256, 0, stream>>>(t0, mb0, mg0, mbe0, mm0, mv0,
      h0b, (float*)nullptr, 256, NGRAPH * 256);
  gemm(h0b, MW1t, t1, NGRAPH, 128, 256);
  bn_relu_epi<<<cdiv(NGRAPH * 128, 256), 256, 0, stream>>>(t1, mb1, mg1, mbe1, mm1, mv1,
      (bf16*)nullptr, h1, 128, NGRAPH * 128);
  final_sigmoid<<<cdiv(NGRAPH, 256), 256, 0, stream>>>(h1, mW2, mb2, (float*)d_out);
}
